// Attention_37477884624910
// MI455X (gfx1250) — matrix operand order repaired and checked by reading, not compiled, not run on hardware
//
#include <hip/hip_runtime.h>
#include <math.h>

// Problem constants (match reference)
#define B_   4
#define S_   2048
#define D_   1024
#define H_   16
#define DK_  64
#define BS_  (B_ * S_)

// CDNA5 WMMA vector types
typedef __attribute__((ext_vector_type(16))) __bf16 v16bf;
typedef __attribute__((ext_vector_type(8)))  __bf16 v8bf;
typedef __attribute__((ext_vector_type(8)))  float  v8f;

static __device__ __forceinline__ v16bf ld_frag(const __bf16* p0, const __bf16* p1) {
    // Two aligned 16B chunks -> one 16-element bf16 fragment register block
    v8bf lo = *(const v8bf*)p0;
    v8bf hi = *(const v8bf*)p1;
    return __builtin_shufflevector(lo, hi, 0,1,2,3,4,5,6,7,8,9,10,11,12,13,14,15);
}

static __device__ __forceinline__ v8f wmma_bf16(v16bf a, v16bf b, v8f c) {
    return __builtin_amdgcn_wmma_f32_16x16x32_bf16(false, a, false, b, (short)0, c, false, false);
}

// ---------------------------------------------------------------------------
// fp32 -> bf16 conversion
// ---------------------------------------------------------------------------
__global__ __launch_bounds__(256) void cvt_bf16_kernel(const float* __restrict__ in,
                                                       __bf16* __restrict__ out, int n) {
    int i = blockIdx.x * 256 + threadIdx.x;
    if (i < n) out[i] = (__bf16)in[i];
}

// ---------------------------------------------------------------------------
// GEMM: P[row, o] = sum_i X[row, i] * W[o, i]   (X: [BS,D] bf16, W: [D,D] bf16)
// Workgroup = 8 waves = 32 rows x 256 cols; each wave: 16 rows x 64 cols.
// mode 0/1: fused RoPE epilogue, store bf16 [B,H,S,DK]  (Q / K)
// mode 2:   store bf16 transposed [B,H,DK,S]            (V)
// mode 3:   store fp32 [BS, D]                          (final Wo projection)
// ---------------------------------------------------------------------------
__global__ __launch_bounds__(256) void gemm16_kernel(const __bf16* __restrict__ X,
                                                     const __bf16* __restrict__ W,
                                                     __bf16* __restrict__ outB,
                                                     float* __restrict__ outF,
                                                     int mode) {
    __shared__ __attribute__((aligned(16))) float ropeLds[8][16][17];

    const int tid  = threadIdx.x;
    const int w    = tid >> 5;        // wave in workgroup (0..7)
    const int l    = tid & 31;        // lane
    const int half = l >> 4;          // lane half (0/1)
    const int ln   = l & 15;          // lane-in-half

    const int rowBase  = blockIdx.x * 32 + ((w & 1) << 4);           // 16-row band
    const int colGroup = blockIdx.y * 256 + ((w >> 1) << 6);         // 64-col band

    v8f acc[4] = {};

    const __bf16* xrow = X + (size_t)(rowBase + ln) * D_;
    for (int k0 = 0; k0 < D_; k0 += 32) {
        // A fragment: 16x32 of X (ISA 16-bit A layout)
        v16bf a = ld_frag(xrow + k0 + 8 * half, xrow + k0 + 16 + 8 * half);
#pragma unroll
        for (int nt = 0; nt < 4; nt++) {
            // B fragment: 32x16 of W^T == a 32-bf16 span of a W row
            const __bf16* wrow = W + (size_t)(colGroup + 16 * nt + ln) * D_ + k0;
            v16bf b = ld_frag(wrow + 8 * half, wrow + 16 + 8 * half);
            acc[nt] = wmma_bf16(a, b, acc[nt]);
        }
    }

    if (mode == 3) {  // output projection -> fp32 d_out
#pragma unroll
        for (int nt = 0; nt < 4; nt++)
#pragma unroll
            for (int r = 0; r < 8; r++) {
                int gRow = rowBase + r + 8 * half;
                outF[(size_t)gRow * D_ + colGroup + 16 * nt + ln] = acc[nt][r];
            }
        return;
    }

    if (mode == 2) {  // V -> transposed [B,H,DK,S] bf16
#pragma unroll
        for (int nt = 0; nt < 4; nt++)
#pragma unroll
            for (int r = 0; r < 8; r++) {
                int gRow = rowBase + r + 8 * half;
                int b = gRow >> 11;           // / S_
                int s = gRow & (S_ - 1);
                int o = colGroup + 16 * nt + ln;
                int h = o >> 6, d = o & 63;
                outB[((size_t)(b * H_ + h) * DK_ + d) * S_ + s] = (__bf16)acc[nt][r];
            }
        return;
    }

    // modes 0/1: fused RoPE, store [B,H,S,DK] bf16
#pragma unroll
    for (int nt = 0; nt < 4; nt++) {
        // spill fp32 16x16 tile to this wave's LDS slab (wave-synchronous)
#pragma unroll
        for (int r = 0; r < 8; r++)
            ropeLds[w][r + 8 * half][ln] = acc[nt][r];
        asm volatile("s_wait_dscnt 0" ::: "memory");
        // 128 rotation pairs; 4 per lane
#pragma unroll
        for (int t = 0; t < 4; t++) {
            int idx = l * 4 + t;
            int row = idx >> 3;   // 0..15
            int p   = idx & 7;    // pair 0..7
            float t1 = ropeLds[w][row][2 * p];
            float t2 = ropeLds[w][row][2 * p + 1];
            int gRow = rowBase + row;
            int b = gRow >> 11;
            int s = gRow & (S_ - 1);
            int o = colGroup + 16 * nt + 2 * p;
            int h = o >> 6, d0 = o & 63;
            float fi  = (float)(d0 >> 1);
            // ang = s * THETA^(-2i/DK),  ln(10000) = 9.210340372
            float ang = (float)s * __expf(fi * (-2.0f / 64.0f) * 9.2103403719761836f);
            float sn, cs;
            __sincosf(ang, &sn, &cs);
            __bf16* dst = outB + ((size_t)(b * H_ + h) * S_ + s) * DK_ + d0;
            dst[0] = (__bf16)(cs * t1 - sn * t2);
            dst[1] = (__bf16)(sn * t1 + cs * t2);
        }
        asm volatile("s_wait_dscnt 0" ::: "memory");
    }
}

// ---------------------------------------------------------------------------
// Causal flash attention. One wave per (b, h, 16-query tile). KV tile = 32.
// Q/K: [B,H,S,DK] bf16 (RoPE applied). Vt: [B,H,DK,S] bf16. O: [BS,D] bf16.
// ---------------------------------------------------------------------------
__global__ __launch_bounds__(256) void fa_kernel(const __bf16* __restrict__ Q,
                                                 const __bf16* __restrict__ K,
                                                 const __bf16* __restrict__ Vt,
                                                 __bf16* __restrict__ O) {
    __shared__ __attribute__((aligned(16))) __bf16 pTile[8][16][48];  // padded rows (96B)

    const int tid  = threadIdx.x;
    const int w    = tid >> 5;
    const int l    = tid & 31;
    const int half = l >> 4;
    const int ln   = l & 15;

    const int gw = blockIdx.x * 8 + w;       // 0..8191
    const int qt = gw & 127;                 // S/16 tiles
    const int h  = (gw >> 7) & 15;
    const int b  = gw >> 11;
    const int q0 = qt << 4;

    const size_t bh = (size_t)(b * H_ + h);
    const __bf16* Kp = K + bh * S_ * DK_;
    const __bf16* Vp = Vt + bh * DK_ * S_;

    // Q A-fragments (16 queries x 64 dims -> two 16x32 fragments)
    v16bf aQ[2];
    {
        const __bf16* qrow = Q + (bh * S_ + q0 + ln) * DK_;
        aQ[0] = ld_frag(qrow + 8 * half,       qrow + 16 + 8 * half);
        aQ[1] = ld_frag(qrow + 32 + 8 * half,  qrow + 48 + 8 * half);
    }

    float mRun[8], lRun[8];
#pragma unroll
    for (int r = 0; r < 8; r++) { mRun[r] = -3.0e38f; lRun[r] = 0.0f; }
    v8f oAcc[4] = {};

    const int nkt = ((q0 + 15) >> 5) + 1;    // causal: only tiles with k0 <= q_max
    for (int kt = 0; kt < nkt; kt++) {
        const int k0 = kt << 5;

        if (kt + 1 < nkt)  // warm L2->L0 for the next K tile
            __builtin_prefetch(Kp + (size_t)(k0 + 32 + l) * DK_, 0, 0);

        // scores: two 16(q) x 16(key) tiles, K-dim = DK split 2x32
        v8f sc[2];
#pragma unroll
        for (int j = 0; j < 2; j++) {
            const __bf16* krow = Kp + (size_t)(k0 + 16 * j + ln) * DK_;
            v16bf b0 = ld_frag(krow + 8 * half,      krow + 16 + 8 * half);
            v16bf b1 = ld_frag(krow + 32 + 8 * half, krow + 48 + 8 * half);
            v8f z = {};
            z     = wmma_bf16(aQ[0], b0, z);
            sc[j] = wmma_bf16(aQ[1], b1, z);
        }

        // masked online softmax (rows live per (r, lane-half); cols across 16 lanes)
        float pv0[8], pv1[8];
#pragma unroll
        for (int r = 0; r < 8; r++) {
            int q = q0 + r + 8 * half;
            float x0 = sc[0][r] * 0.125f;   // 1/sqrt(64)
            float x1 = sc[1][r] * 0.125f;
            if (k0 + ln      > q) x0 = -3.0e38f;
            if (k0 + 16 + ln > q) x1 = -3.0e38f;
            float mx = fmaxf(x0, x1);
#pragma unroll
            for (int off = 1; off < 16; off <<= 1)
                mx = fmaxf(mx, __shfl_xor(mx, off, 32));
            float mNew = fmaxf(mRun[r], mx);
            float p0 = __expf(x0 - mNew);
            float p1 = __expf(x1 - mNew);
            float rs = p0 + p1;
#pragma unroll
            for (int off = 1; off < 16; off <<= 1)
                rs += __shfl_xor(rs, off, 32);
            float corr = __expf(mRun[r] - mNew);
            lRun[r] = lRun[r] * corr + rs;
            mRun[r] = mNew;
            pv0[r] = p0; pv1[r] = p1;
#pragma unroll
            for (int f = 0; f < 4; f++) oAcc[f][r] *= corr;
        }

        // P (C layout) -> LDS -> A layout bf16 fragment
#pragma unroll
        for (int r = 0; r < 8; r++) {
            pTile[w][r + 8 * half][ln]      = (__bf16)pv0[r];
            pTile[w][r + 8 * half][16 + ln] = (__bf16)pv1[r];
        }
        asm volatile("s_wait_dscnt 0" ::: "memory");
        const __bf16* prow = &pTile[w][ln][0];
        v16bf aP = ld_frag(prow + 8 * half, prow + 16 + 8 * half);

        // O += P[16x32] * V[32x64]; Vt rows are contiguous along keys
#pragma unroll
        for (int f = 0; f < 4; f++) {
            const __bf16* vrow = Vp + (size_t)(16 * f + ln) * S_ + k0;
            v16bf bv = ld_frag(vrow + 8 * half, vrow + 16 + 8 * half);
            oAcc[f] = wmma_bf16(aP, bv, oAcc[f]);
        }
        asm volatile("s_wait_dscnt 0" ::: "memory");  // WAR: next iter rewrites pTile
    }

    // normalize and store to O [BS, D] bf16
#pragma unroll
    for (int f = 0; f < 4; f++)
#pragma unroll
        for (int r = 0; r < 8; r++) {
            int q = q0 + r + 8 * half;
            float val = oAcc[f][r] / lRun[r];
            O[((size_t)b * S_ + q) * D_ + h * DK_ + 16 * f + ln] = (__bf16)val;
        }
}

// ---------------------------------------------------------------------------
// Host launcher
// ---------------------------------------------------------------------------
extern "C" void kernel_launch(void* const* d_in, const int* in_sizes, int n_in,
                              void* d_out, int out_size, void* d_ws, size_t ws_size,
                              hipStream_t stream) {
    const float* x  = (const float*)d_in[0];
    const float* Wq = (const float*)d_in[1];
    const float* Wk = (const float*)d_in[2];
    const float* Wv = (const float*)d_in[3];
    const float* Wo = (const float*)d_in[4];
    float* out = (float*)d_out;

    const size_t nX = (size_t)BS_ * D_;   // 8,388,608
    const size_t nW = (size_t)D_ * D_;    // 1,048,576

    char* ws = (char*)d_ws;
    __bf16* xb  = (__bf16*)ws;            ws += nX * sizeof(__bf16);
    __bf16* Wqb = (__bf16*)ws;            ws += nW * sizeof(__bf16);
    __bf16* Wkb = (__bf16*)ws;            ws += nW * sizeof(__bf16);
    __bf16* Wvb = (__bf16*)ws;            ws += nW * sizeof(__bf16);
    __bf16* Wob = (__bf16*)ws;            ws += nW * sizeof(__bf16);
    __bf16* Qb  = (__bf16*)ws;            ws += nX * sizeof(__bf16);
    __bf16* Kb  = (__bf16*)ws;            ws += nX * sizeof(__bf16);
    __bf16* Vtb = (__bf16*)ws;            ws += nX * sizeof(__bf16);
    __bf16* Ob  = (__bf16*)ws;            ws += nX * sizeof(__bf16);

    // fp32 -> bf16
    cvt_bf16_kernel<<<(int)((nX + 255) / 256), 256, 0, stream>>>(x,  xb,  (int)nX);
    cvt_bf16_kernel<<<(int)((nW + 255) / 256), 256, 0, stream>>>(Wq, Wqb, (int)nW);
    cvt_bf16_kernel<<<(int)((nW + 255) / 256), 256, 0, stream>>>(Wk, Wkb, (int)nW);
    cvt_bf16_kernel<<<(int)((nW + 255) / 256), 256, 0, stream>>>(Wv, Wvb, (int)nW);
    cvt_bf16_kernel<<<(int)((nW + 255) / 256), 256, 0, stream>>>(Wo, Wob, (int)nW);

    dim3 gGemm(BS_ / 32, D_ / 256);   // 256 x 4 workgroups
    gemm16_kernel<<<gGemm, 256, 0, stream>>>(xb, Wqb, Qb,  nullptr, 0);  // Q + RoPE
    gemm16_kernel<<<gGemm, 256, 0, stream>>>(xb, Wkb, Kb,  nullptr, 1);  // K + RoPE
    gemm16_kernel<<<gGemm, 256, 0, stream>>>(xb, Wvb, Vtb, nullptr, 2);  // V transposed

    fa_kernel<<<(B_ * H_ * (S_ / 16)) / 8, 256, 0, stream>>>(Qb, Kb, Vtb, Ob);

    gemm16_kernel<<<gGemm, 256, 0, stream>>>(Ob, Wob, nullptr, out, 3);  // Wo -> fp32
}
